// CrossAttention_5987184410793
// MI455X (gfx1250) — matrix operand order repaired and checked by reading, not compiled, not run on hardware
//
#include <hip/hip_runtime.h>
#include <hip/hip_bf16.h>

// ---------------------------------------------------------------------------
// Point-cloud local attention (B=8, C=64, N=2048, K=32, H=8, D=8)
//   q = Wq*pcd ; k = Wk*nbr ; v = Wv*nbr ; per-point softmax over 32 neighbors
// Per block: 2 points end-to-end.
//   - Neighbor tile staged via GLOBAL_LOAD_ASYNC_TO_LDS_B128 (ASYNCcnt),
//     overlapped with weight staging VALU work
//   - Channel projections via v_wmma_f32_16x16x32_bf16 (f32 accumulate)
//   - Tiny softmax/contraction in VALU with wave shuffles
// ---------------------------------------------------------------------------

typedef __attribute__((ext_vector_type(16))) __bf16 v16bf;
typedef __attribute__((ext_vector_type(8)))  float  v8f;

namespace {
constexpr int Bc  = 8;     // batches
constexpr int Cc  = 64;    // channels
constexpr int Nc  = 2048;  // points
constexpr int KN  = 32;    // neighbors per point
constexpr int Hh  = 8;     // heads
constexpr int TP  = 2;     // points per block
constexpr int COLS = TP * KN;        // 64 neighbor columns per block
constexpr int SW  = 68;    // LDS row stride (shorts) for weights (bank-skewed)
constexpr int SX  = 68;    // LDS row stride (shorts) for X^T / P^T
constexpr int SKV = 66;    // LDS row stride (shorts) for Kproj/Vproj (bf16)
}

__device__ __forceinline__ unsigned short f2bf(float f) {
  unsigned u = __float_as_uint(f);
  u += 0x7FFFu + ((u >> 16) & 1u);          // round-to-nearest-even
  return (unsigned short)(u >> 16);
}
__device__ __forceinline__ float bf2f(unsigned short h) {
  return __uint_as_float(((unsigned)h) << 16);
}

union FragAB { v16bf v; unsigned u[8]; };
union FragC  { v8f   v; float    f[8]; };

__global__ __launch_bounds__(256, 1)
void pct_local_attn(const float* __restrict__ pcd,
                    const float* __restrict__ nbr,
                    const float* __restrict__ Wq,
                    const float* __restrict__ Wk,
                    const float* __restrict__ Wv,
                    float* __restrict__ out) {
  __shared__ unsigned short sWk[64 * SW];
  __shared__ unsigned short sWv[64 * SW];
  __shared__ unsigned short sWq[64 * SW];
  __shared__ unsigned short sXt[COLS * SX];   // neighbors^T: [col][ci] (bf16)
  __shared__ unsigned short sPt[16 * SX];     // pcd^T (cols 0..TP-1 valid)
  // Kproj/Vproj bf16 region; ALSO aliased as the fp32 async staging buffer
  // during phase 1 (16896 B >= 16384 B needed for 128 rows x 32 f32).
  __shared__ __align__(16) unsigned short sKVraw[2 * 64 * SKV];
  __shared__ float sQ[64 * TP];               // Qproj f32  [outCh][point]
  __shared__ float sAtt[TP * Hh * KN];        // softmax weights
  __shared__ float sOut[64 * TP];             // staged output

  unsigned short* sKp = sKVraw;
  unsigned short* sVp = sKVraw + 64 * SKV;
  float* sNstage = (float*)sKVraw;            // phase-1 alias (fp32 rows)

  const int t    = threadIdx.x;
  const int lane = t & 31;
  const int wave = t >> 5;
  const int bid  = blockIdx.x;
  const int b    = bid / (Nc / TP);
  const int n0   = (bid % (Nc / TP)) * TP;

  // ---- Phase 1a: kick off async HBM->LDS copies of the neighbor tile ------
  // 128 rows of 128 B each; each thread issues one B128 copy per iteration.
  // Tracked by ASYNCcnt, so the wave keeps running while data is in flight.
#pragma unroll
  for (int it = 0; it < (TP * Cc) / 32; ++it) {   // 4 iterations
    int r = (t >> 3) + 32 * it;                   // 0..127  (p,c) row
    int p = r >> 6, c = r & 63;
    int j = t & 7;
    const float* gsrc = nbr + (((size_t)b * Cc + c) * Nc + (n0 + p)) * KN + j * 4;
    unsigned ldsoff = (unsigned)(size_t)&sNstage[r * 32 + j * 4];
    asm volatile("global_load_async_to_lds_b128 %0, %1, off"
                 :: "v"(ldsoff), "v"(gsrc) : "memory");
  }

  // ---- Phase 1b: stage weights (fp32 -> bf16) while copies are in flight ---
  {
    int row = t >> 2;             // 0..63
    int cs  = (t & 3) * 16;       // 0,16,32,48
    const float4* q4 = (const float4*)(Wq + row * Cc + cs);
    const float4* k4 = (const float4*)(Wk + row * Cc + cs);
    const float4* v4 = (const float4*)(Wv + row * Cc + cs);
#pragma unroll
    for (int i = 0; i < 4; ++i) {
      float4 a = q4[i], bb = k4[i], cc = v4[i];
      int o = row * SW + cs + i * 4;
      sWq[o+0]=f2bf(a.x);  sWq[o+1]=f2bf(a.y);  sWq[o+2]=f2bf(a.z);  sWq[o+3]=f2bf(a.w);
      sWk[o+0]=f2bf(bb.x); sWk[o+1]=f2bf(bb.y); sWk[o+2]=f2bf(bb.z); sWk[o+3]=f2bf(bb.w);
      sWv[o+0]=f2bf(cc.x); sWv[o+1]=f2bf(cc.y); sWv[o+2]=f2bf(cc.z); sWv[o+3]=f2bf(cc.w);
    }
  }
  // pcd tile (column-major, padded to 16 cols with zeros)
  for (int i = t; i < 16 * SX; i += 256) sPt[i] = 0;
  if (t < Cc * TP) {
    int c = t >> 1, p = t & 1;
    float v = pcd[((size_t)b * Cc + c) * Nc + n0 + p];
    sPt[p * SX + c] = f2bf(v);
  }

  // ---- Phase 1c: wait for async copies, then transpose+convert to bf16 ----
  // Each thread converts exactly the rows it copied, so waiting on this
  // wave's ASYNCcnt is sufficient before the LDS reads.
  asm volatile("s_wait_asynccnt 0x0" ::: "memory");
#pragma unroll
  for (int it = 0; it < (TP * Cc) / 32; ++it) {
    int r = (t >> 3) + 32 * it;
    int p = r >> 6, c = r & 63;
    int j = t & 7;
    float4 val = *(const float4*)&sNstage[r * 32 + j * 4];
    int cb = p * KN + j * 4;
    sXt[(cb + 0) * SX + c] = f2bf(val.x);
    sXt[(cb + 1) * SX + c] = f2bf(val.y);
    sXt[(cb + 2) * SX + c] = f2bf(val.z);
    sXt[(cb + 3) * SX + c] = f2bf(val.w);
  }
  __syncthreads();   // sXt complete; sNstage dead -> sKp/sVp may be written

  // ---- Phase 2: projections via WMMA bf16 ----------------------------------
  // Tiles: Kproj 4x4=16, Vproj 16, Qproj 4 (1 padded col tile). 36 total.
  const int nTK = 4 * (COLS / 16);   // 16
  const int nTiles = 2 * nTK + 4;    // 36
  const int m15 = lane & 15;
  const int hi  = lane >> 4;         // 0|1
  for (int tile = wave; tile < nTiles; tile += 8) {
    const unsigned short* Wl;
    const unsigned short* Bsrc;
    unsigned short* dstKV = nullptr;
    int mt, ct;
    if (tile < nTK)            { Wl = sWk; dstKV = sKp; mt = tile >> 2;         ct = tile & 3;          Bsrc = sXt; }
    else if (tile < 2 * nTK)   { Wl = sWv; dstKV = sVp; mt = (tile - nTK) >> 2; ct = (tile - nTK) & 3;  Bsrc = sXt; }
    else                       { Wl = sWq; dstKV = nullptr; mt = tile - 2 * nTK; ct = 0;                Bsrc = sPt; }

    FragC acc; acc.v = (v8f){0.f,0.f,0.f,0.f,0.f,0.f,0.f,0.f};
#pragma unroll
    for (int ks = 0; ks < 2; ++ks) {            // 64-deep channel dim, 2 x K=32
      FragAB A, Bf;
      int m = mt * 16 + m15;                    // A row (out channel)
      int col = ct * 16 + m15;                  // B col (neighbor column)
#pragma unroll
      for (int v = 0; v < 8; ++v) {
        int ka = (v < 4 ? 0 : 16) + (hi ? 8 : 0) + 2 * (v & 3) + ks * 32;  // ISA A layout
        A.u[v] = *(const unsigned*)&Wl[m * SW + ka];
        int kb = (v < 4 ? 0 : 16) + (hi ? 8 : 0) + 2 * (v & 3) + ks * 32;  // ISA B layout
        Bf.u[v] = *(const unsigned*)&Bsrc[col * SX + kb];
      }
      acc.v = __builtin_amdgcn_wmma_f32_16x16x32_bf16(
          false, A.v, false, Bf.v, (short)0, acc.v, false, false);
    }
    // D layout: lane -> col (lane&15); VGPR v -> row mt*16 + v + (hi?8:0)
    int rbase = mt * 16 + (hi ? 8 : 0);
    if (dstKV) {
      int col = ct * 16 + m15;
#pragma unroll
      for (int v = 0; v < 8; ++v) dstKV[(rbase + v) * SKV + col] = f2bf(acc.f[v]);
    } else if (m15 < TP) {
#pragma unroll
      for (int v = 0; v < 8; ++v) sQ[(rbase + v) * TP + m15] = acc.f[v];
    }
  }
  __syncthreads();

  // ---- Phase 3: softmax attention (8 lanes per (point, head) pair) ---------
  if (t < TP * Hh * 8) {          // 128 threads
    int pair = t >> 3;            // 0..15
    int j    = t & 7;
    int p    = pair >> 3;         // point 0..1
    int h    = pair & 7;          // head
    const float scale = 0.35355339059327373f;   // 1/sqrt(D=8)
    float q[8];
#pragma unroll
    for (int d = 0; d < 8; ++d) q[d] = sQ[(h * 8 + d) * TP + p];
    float e[4];
#pragma unroll
    for (int kk = 0; kk < 4; ++kk) {
      int k = j + 8 * kk;
      float s = 0.f;
#pragma unroll
      for (int d = 0; d < 8; ++d)
        s += q[d] * bf2f(sKp[(h * 8 + d) * SKV + p * KN + k]);
      e[kk] = s * scale;
    }
    float m = fmaxf(fmaxf(e[0], e[1]), fmaxf(e[2], e[3]));
    for (int msk = 1; msk < 8; msk <<= 1) m = fmaxf(m, __shfl_xor(m, msk, 32));
    float sum = 0.f;
#pragma unroll
    for (int kk = 0; kk < 4; ++kk) { e[kk] = __expf(e[kk] - m); sum += e[kk]; }
    for (int msk = 1; msk < 8; msk <<= 1) sum += __shfl_xor(sum, msk, 32);
    float inv = 1.f / sum;
#pragma unroll
    for (int kk = 0; kk < 4; ++kk) sAtt[pair * KN + j + 8 * kk] = e[kk] * inv;
    // same-wave LDS ops are in-order: safe to read back without a barrier
    float o = 0.f;
#pragma unroll
    for (int k = 0; k < KN; ++k)
      o += sAtt[pair * KN + k] * bf2f(sVp[(h * 8 + j) * SKV + p * KN + k]);
    sOut[(h * 8 + j) * TP + p] = o;
  }
  __syncthreads();

  // ---- Phase 4: output [B,C,N] ---------------------------------------------
  if (t < Cc * TP) {
    int c = t >> 1, pp = t & 1;
    out[((size_t)b * Cc + c) * Nc + n0 + pp] = sOut[c * TP + pp];
  }
}

extern "C" void kernel_launch(void* const* d_in, const int* in_sizes, int n_in,
                              void* d_out, int out_size, void* d_ws, size_t ws_size,
                              hipStream_t stream) {
  (void)in_sizes; (void)n_in; (void)out_size; (void)d_ws; (void)ws_size;
  const float* pcd = (const float*)d_in[0];
  const float* nbr = (const float*)d_in[1];
  const float* Wq  = (const float*)d_in[2];
  const float* Wk  = (const float*)d_in[3];
  const float* Wv  = (const float*)d_in[4];
  float* out = (float*)d_out;
  dim3 grid(Bc * (Nc / TP));   // 8192 blocks
  pct_local_attn<<<grid, 256, 0, stream>>>(pcd, nbr, Wq, Wk, Wv, out);
}
